// ProtTopkPool_65360812310549
// MI455X (gfx1250) — matrix operand order repaired and checked by reading, not compiled, not run on hardware
//
#include <hip/hip_runtime.h>
#include <hip/hip_bf16.h>

#define DIM_   128
#define NFEAT  14
#define KPAD1  32
#define B_     128
#define NPER   512
#define DEG_   8
#define NNODES (B_ * NPER)
#define NEDGE  (NNODES * DEG_)

typedef __attribute__((ext_vector_type(16))) _Float16 v16h;
typedef __attribute__((ext_vector_type(8)))  float    v8f;
typedef __attribute__((ext_vector_type(4)))  float    v4f;

// ---------------- zero fill ----------------
__global__ void k_zero(float* __restrict__ p, long long n) {
    long long i = (long long)blockIdx.x * blockDim.x + threadIdx.x;
    if (i < n) p[i] = 0.0f;
}

// ---------------- pad-copy x: [M x 14] -> [M x 32] (zero padded) ----------------
__global__ void k_padx(const float* __restrict__ x, float* __restrict__ xp, int M) {
    int i = blockIdx.x * blockDim.x + threadIdx.x;
    if (i >= M * KPAD1) return;
    int r = i >> 5, f = i & 31;
    xp[i] = (f < NFEAT) ? x[r * NFEAT + f] : 0.0f;
}

// ---------------- edge scatter-add: msg[dst*dstride+f] += x[src*sstride+f] ------
__global__ void k_scatter(const float* __restrict__ x, const int* __restrict__ src,
                          const int* __restrict__ dst, const int* __restrict__ mask,
                          float* __restrict__ msg, int K, int sstride, int dstride) {
    int e = blockIdx.x;
    int f = threadIdx.x;
    if (f >= K) return;
    if (mask && mask[e] == 0) return;
    atomicAdd(&msg[(size_t)dst[e] * dstride + f], x[(size_t)src[e] * sstride + f]);
}

// ---------------- pack weight [K x Nn] f32 -> WMMA B fragments f16 ----------------
// layout: P[((chunk*NT + ntile)*32 + lane)*16 + e], k = chunk*32 + 8*(lane/16) + e (+8 if e >= 8)
__global__ void k_packB(const float* __restrict__ W, _Float16* __restrict__ P,
                        int K, int Nn) {
    const int chunk = blockIdx.x;
    const int nt    = blockIdx.y;
    const int lane  = threadIdx.x;     // 32
    const int NT    = Nn >> 4;
    const int col   = nt * 16 + (lane & 15);
    const int kb    = chunk * 32 + ((lane >= 16) ? 8 : 0);
    v16h vals;
#pragma unroll
    for (int e = 0; e < 16; ++e) {
        int k = kb + e + ((e >= 8) ? 8 : 0);
        vals[e] = (k < K) ? (_Float16)W[(size_t)k * Nn + col] : (_Float16)0.0f;
    }
    *(v16h*)(P + ((size_t)(chunk * NT + nt) * 32 + lane) * 16) = vals;
}

// ---------------- A fragment: 16x32 f16, from f32 row-major [ldA == K] ----------
template<int K>
__device__ __forceinline__ v16h load_a_frag(const float* __restrict__ A,
                                            int row, int kk, bool hi) {
    const float* p = A + (size_t)row * K + kk + (hi ? 8 : 0);
    v4f f0 = *(const v4f*)(p);
    v4f f1 = *(const v4f*)(p + 4);
    v4f f2 = *(const v4f*)(p + 16);
    v4f f3 = *(const v4f*)(p + 20);
    v16h a;
#pragma unroll
    for (int j = 0; j < 4; ++j) {
        a[j]      = (_Float16)f0[j];
        a[4 + j]  = (_Float16)f1[j];
        a[8 + j]  = (_Float16)f2[j];
        a[12 + j] = (_Float16)f3[j];
    }
    return a;
}

// ---------------- WMMA GEMM: C[M x NT*16] = relu?(A1*B1 (+A2*B2) + bias) --------
// One wave per MT*16-row strip; each B fragment feeds MT WMMAs.
// launch_bounds(32,1): single-wave block, allow full VGPR budget (no spills).
template<int K, int NT, int MT, bool HASA2>
__global__ void __launch_bounds__(32, 1)
k_gemm(const float* __restrict__ A1, const _Float16* __restrict__ Bp1,
       const float* __restrict__ A2, const _Float16* __restrict__ Bp2,
       const float* __restrict__ bias, float* __restrict__ C, int relu) {
    constexpr int Nn = NT * 16;
    constexpr int NC = K / 32;
    constexpr int NPASS = HASA2 ? 2 : 1;
    const int lane = threadIdx.x;               // wave32
    const int m0   = blockIdx.x * (MT * 16);
    const bool hi  = lane >= 16;
    const int l15  = lane & 15;

    v8f acc[MT][NT];
#pragma unroll
    for (int mt = 0; mt < MT; ++mt)
#pragma unroll
        for (int nt = 0; nt < NT; ++nt) acc[mt][nt] = v8f{};

#pragma unroll
    for (int pass = 0; pass < NPASS; ++pass) {
        const float*    A  = (pass == 0) ? A1  : A2;
        const _Float16* Bp = (pass == 0) ? Bp1 : Bp2;
#pragma unroll
        for (int c = 0; c < NC; ++c) {
            v16h a[MT];
#pragma unroll
            for (int mt = 0; mt < MT; ++mt)
                a[mt] = load_a_frag<K>(A, m0 + mt * 16 + l15, c * 32, hi);
            const _Float16* bb = Bp + ((size_t)c * NT * 32 + lane) * 16;
#pragma unroll
            for (int nt = 0; nt < NT; ++nt) {
                v16h b = *(const v16h*)(bb + (size_t)nt * 32 * 16);
#pragma unroll
                for (int mt = 0; mt < MT; ++mt)
                    acc[mt][nt] = __builtin_amdgcn_wmma_f32_16x16x32_f16(
                        false, a[mt], false, b, (short)0, acc[mt][nt], false, false);
            }
        }
    }

#pragma unroll
    for (int mt = 0; mt < MT; ++mt)
#pragma unroll
        for (int nt = 0; nt < NT; ++nt) {
            const int col = nt * 16 + l15;
            const float bv = bias[col];
#pragma unroll
            for (int v = 0; v < 8; ++v) {
                int r = m0 + mt * 16 + v + (hi ? 8 : 0);
                float val = acc[mt][nt][v] + bv;
                if (relu) val = fmaxf(val, 0.0f);
                C[(size_t)r * Nn + col] = val;
            }
        }
}

// ---------------- 1/||w|| ----------------
__global__ void k_invnorm(const float* __restrict__ w, float* __restrict__ outv) {
    __shared__ float red[DIM_];
    int t = threadIdx.x;
    float v = w[t];
    red[t] = v * v;
    __syncthreads();
    for (int s = DIM_ / 2; s > 0; s >>= 1) {
        if (t < s) red[t] += red[t + s];
        __syncthreads();
    }
    if (t == 0) outv[0] = 1.0f / sqrtf(red[0]);
}

// ---------------- score = tanh(x.w / ||w||) ----------------
__global__ void k_score(const float* __restrict__ x, const float* __restrict__ w,
                        const float* __restrict__ invnorm, float* __restrict__ score, int M) {
    int i = blockIdx.x * blockDim.x + threadIdx.x;
    if (i >= M) return;
    float s = 0.0f;
    for (int f = 0; f < DIM_; ++f) s += x[(size_t)i * DIM_ + f] * w[f];
    score[i] = tanhf(s * invnorm[0]);
}

// ---------------- per-graph top-k (odd-even sort in LDS, descending, stable) ----
__global__ void k_topk(const float* __restrict__ x, const float* __restrict__ score,
                       float* __restrict__ xo, int* __restrict__ new_idx, int n, int k) {
    __shared__ float s[NPER];
    __shared__ int   id[NPER];
    int g = blockIdx.x;
    int t = threadIdx.x;                 // blockDim == NPER
    s[t]  = (t < n) ? score[g * n + t] : -3.0e38f;
    id[t] = t;
    __syncthreads();
    for (int it = 0; it < n; ++it) {
        int i = 2 * t + (it & 1);
        if (i + 1 < n) {
            if (s[i] < s[i + 1]) {
                float ts = s[i]; s[i] = s[i + 1]; s[i + 1] = ts;
                int   ti = id[i]; id[i] = id[i + 1]; id[i + 1] = ti;
            }
        }
        __syncthreads();
    }
    if (t < n) new_idx[g * n + id[t]] = (t < k) ? (g * k + t) : -1;
    if (t < k) {
        float sc = s[t];
        size_t srow = (size_t)(g * n + id[t]) * DIM_;
        size_t drow = (size_t)(g * k + t) * DIM_;
        for (int f = 0; f < DIM_; ++f) xo[drow + f] = x[srow + f] * sc;
    }
}

// ---------------- remap edges through new_idx ----------------
__global__ void k_remap(const int* __restrict__ src, const int* __restrict__ dst,
                        const int* __restrict__ mask, const int* __restrict__ new_idx,
                        int* __restrict__ so, int* __restrict__ d_o, int* __restrict__ mo,
                        int E) {
    int e = blockIdx.x * blockDim.x + threadIdx.x;
    if (e >= E) return;
    int m  = mask ? mask[e] : 1;
    int s2 = new_idx[src[e]];
    int d2 = new_idx[dst[e]];
    int valid = (m != 0) && (s2 >= 0) && (d2 >= 0);
    so[e]  = valid ? s2 : 0;
    d_o[e] = valid ? d2 : 0;
    mo[e]  = valid ? 1 : 0;
}

// ---------------- readout: z += [max ; mean] over k nodes per graph ----------------
__global__ void k_readout(const float* __restrict__ x, float* __restrict__ z, int k, int first) {
    int g = blockIdx.x, f = threadIdx.x;   // blockDim == 128
    float mx = -3.0e38f, sum = 0.0f;
    for (int j = 0; j < k; ++j) {
        float v = x[(size_t)(g * k + j) * DIM_ + f];
        mx  = fmaxf(mx, v);
        sum += v;
    }
    float mean = sum / (float)k;
    if (first) { z[g * 256 + f] = mx;  z[g * 256 + DIM_ + f] = mean; }
    else       { z[g * 256 + f] += mx; z[g * 256 + DIM_ + f] += mean; }
}

// ---------------- final linear (64 -> 1) ----------------
__global__ void k_lin3(const float* __restrict__ z, const float* __restrict__ w,
                       const float* __restrict__ b, float* __restrict__ out) {
    int g = blockIdx.x * blockDim.x + threadIdx.x;
    if (g >= B_) return;
    float s = b[0];
    for (int f = 0; f < 64; ++f) s += z[g * 64 + f] * w[f];
    out[g] = s;
}

extern "C" void kernel_launch(void* const* d_in, const int* in_sizes, int n_in,
                              void* d_out, int out_size, void* d_ws, size_t ws_size,
                              hipStream_t stream) {
    (void)in_sizes; (void)n_in; (void)out_size; (void)ws_size;
    const float* x      = (const float*)d_in[0];
    const float* c1r    = (const float*)d_in[1];
    const float* c1root = (const float*)d_in[2];
    const float* c1b    = (const float*)d_in[3];
    const float* c2r    = (const float*)d_in[4];
    const float* c2root = (const float*)d_in[5];
    const float* c2b    = (const float*)d_in[6];
    const float* c3r    = (const float*)d_in[7];
    const float* c3root = (const float*)d_in[8];
    const float* c3b    = (const float*)d_in[9];
    const float* p1w    = (const float*)d_in[10];
    const float* p2w    = (const float*)d_in[11];
    const float* p3w    = (const float*)d_in[12];
    const float* l1w    = (const float*)d_in[13];
    const float* l1b    = (const float*)d_in[14];
    const float* l2w    = (const float*)d_in[15];
    const float* l2b    = (const float*)d_in[16];
    const float* l3w    = (const float*)d_in[17];
    const float* l3b    = (const float*)d_in[18];
    const int*   ei     = (const int*)d_in[19];   // [2, E] : src then dst
    float* out = (float*)d_out;

    const int k1 = (4 * NPER) / 5;
    const int k2 = (4 * k1) / 5;
    const int k3 = (4 * k2) / 5;
    const int M0 = NNODES;
    const int M1 = B_ * k1;
    const int M2 = B_ * k2;

    // workspace partition (256-byte aligned blocks)
    char* wp = (char*)d_ws;
    auto alloc = [&](size_t bytes) -> char* {
        char* r = wp;
        wp += (bytes + 255) & ~(size_t)255;
        return r;
    };
    float* bufC  = (float*)alloc((size_t)M0 * DIM_ * 4);
    float* bufA  = (float*)alloc((size_t)M0 * DIM_ * 4);
    float* msg   = (float*)alloc((size_t)M0 * DIM_ * 4);
    float* xpad  = (float*)alloc((size_t)M0 * KPAD1 * 4);
    float* score = (float*)alloc((size_t)M0 * 4);
    float* scal  = (float*)alloc(256);
    int*   nidx  = (int*)alloc((size_t)M0 * 4);
    int* sA = (int*)alloc((size_t)NEDGE * 4);
    int* dA = (int*)alloc((size_t)NEDGE * 4);
    int* mA = (int*)alloc((size_t)NEDGE * 4);
    int* sB = (int*)alloc((size_t)NEDGE * 4);
    int* dB = (int*)alloc((size_t)NEDGE * 4);
    int* mB = (int*)alloc((size_t)NEDGE * 4);
    float* z  = (float*)alloc((size_t)B_ * 256 * 4);
    float* t1 = (float*)alloc((size_t)B_ * DIM_ * 4);
    float* t2 = (float*)alloc((size_t)B_ * 64 * 4);
    // packed f16 weight fragments: chunks * NT * 32 lanes * 16 halves
    auto packBytes = [](int Kp, int Nn) { return (size_t)(Kp / 32) * (Nn / 16) * 32 * 16 * 2; };
    _Float16* Pc1r    = (_Float16*)alloc(packBytes(KPAD1, DIM_));
    _Float16* Pc1root = (_Float16*)alloc(packBytes(KPAD1, DIM_));
    _Float16* Pc2r    = (_Float16*)alloc(packBytes(DIM_, DIM_));
    _Float16* Pc2root = (_Float16*)alloc(packBytes(DIM_, DIM_));
    _Float16* Pc3r    = (_Float16*)alloc(packBytes(DIM_, DIM_));
    _Float16* Pc3root = (_Float16*)alloc(packBytes(DIM_, DIM_));
    _Float16* Pl1     = (_Float16*)alloc(packBytes(256, DIM_));
    _Float16* Pl2     = (_Float16*)alloc(packBytes(DIM_, 64));

    // ---- pack all weights into WMMA fragment layout (tiny, runs each launch) ----
    k_packB<<<dim3(KPAD1 / 32, DIM_ / 16), 32, 0, stream>>>(c1r,    Pc1r,    NFEAT, DIM_);
    k_packB<<<dim3(KPAD1 / 32, DIM_ / 16), 32, 0, stream>>>(c1root, Pc1root, NFEAT, DIM_);
    k_packB<<<dim3(DIM_ / 32, DIM_ / 16), 32, 0, stream>>>(c2r,    Pc2r,    DIM_, DIM_);
    k_packB<<<dim3(DIM_ / 32, DIM_ / 16), 32, 0, stream>>>(c2root, Pc2root, DIM_, DIM_);
    k_packB<<<dim3(DIM_ / 32, DIM_ / 16), 32, 0, stream>>>(c3r,    Pc3r,    DIM_, DIM_);
    k_packB<<<dim3(DIM_ / 32, DIM_ / 16), 32, 0, stream>>>(c3root, Pc3root, DIM_, DIM_);
    k_packB<<<dim3(256 / 32, DIM_ / 16), 32, 0, stream>>>(l1w, Pl1, 256, DIM_);
    k_packB<<<dim3(DIM_ / 32, 64 / 16), 32, 0, stream>>>(l2w, Pl2, DIM_, 64);

    // ---- conv1 (K padded 14 -> 32) ----
    k_padx<<<(M0 * KPAD1 + 255) / 256, 256, 0, stream>>>(x, xpad, M0);
    k_zero<<<(int)(((long long)M0 * KPAD1 + 255) / 256), 256, 0, stream>>>(msg, (long long)M0 * KPAD1);
    k_scatter<<<NEDGE, 32, 0, stream>>>(x, ei, ei + NEDGE, nullptr, msg, NFEAT, NFEAT, KPAD1);
    k_gemm<KPAD1, 8, 2, true><<<M0 / 32, 32, 0, stream>>>(msg, Pc1r, xpad, Pc1root, c1b, bufC, 1);
    // ---- pool1 ----
    k_invnorm<<<1, DIM_, 0, stream>>>(p1w, scal);
    k_score<<<(M0 + 127) / 128, 128, 0, stream>>>(bufC, p1w, scal, score, M0);
    k_topk<<<B_, NPER, 0, stream>>>(bufC, score, bufA, nidx, NPER, k1);
    k_remap<<<(NEDGE + 255) / 256, 256, 0, stream>>>(ei, ei + NEDGE, nullptr, nidx, sA, dA, mA, NEDGE);
    k_readout<<<B_, DIM_, 0, stream>>>(bufA, z, k1, 1);

    // ---- conv2 ----
    k_zero<<<(int)(((long long)M1 * DIM_ + 255) / 256), 256, 0, stream>>>(msg, (long long)M1 * DIM_);
    k_scatter<<<NEDGE, DIM_, 0, stream>>>(bufA, sA, dA, mA, msg, DIM_, DIM_, DIM_);
    k_gemm<DIM_, 8, 2, true><<<M1 / 32, 32, 0, stream>>>(msg, Pc2r, bufA, Pc2root, c2b, bufC, 1);
    // ---- pool2 ----
    k_invnorm<<<1, DIM_, 0, stream>>>(p2w, scal);
    k_score<<<(M1 + 127) / 128, 128, 0, stream>>>(bufC, p2w, scal, score, M1);
    k_topk<<<B_, NPER, 0, stream>>>(bufC, score, bufA, nidx, k1, k2);
    k_remap<<<(NEDGE + 255) / 256, 256, 0, stream>>>(sA, dA, mA, nidx, sB, dB, mB, NEDGE);
    k_readout<<<B_, DIM_, 0, stream>>>(bufA, z, k2, 0);

    // ---- conv3 ----
    k_zero<<<(int)(((long long)M2 * DIM_ + 255) / 256), 256, 0, stream>>>(msg, (long long)M2 * DIM_);
    k_scatter<<<NEDGE, DIM_, 0, stream>>>(bufA, sB, dB, mB, msg, DIM_, DIM_, DIM_);
    k_gemm<DIM_, 8, 2, true><<<M2 / 32, 32, 0, stream>>>(msg, Pc3r, bufA, Pc3root, c3b, bufC, 1);
    // ---- pool3 ----
    k_invnorm<<<1, DIM_, 0, stream>>>(p3w, scal);
    k_score<<<(M2 + 127) / 128, 128, 0, stream>>>(bufC, p3w, scal, score, M2);
    k_topk<<<B_, NPER, 0, stream>>>(bufC, score, bufA, nidx, k2, k3);
    k_readout<<<B_, DIM_, 0, stream>>>(bufA, z, k3, 0);

    // ---- MLP head ----
    k_gemm<256, 8, 2, false><<<B_ / 32, 32, 0, stream>>>(z, Pl1, nullptr, nullptr, l1b, t1, 1);
    k_gemm<DIM_, 4, 2, false><<<B_ / 32, 32, 0, stream>>>(t1, Pl2, nullptr, nullptr, l2b, t2, 1);
    k_lin3<<<1, B_, 0, stream>>>(t2, l3w, l3b, out);
}
